// CoTrackerThreeOnline_33973191312017
// MI455X (gfx1250) — matrix operand order repaired and checked by reading, not compiled, not run on hardware
//
#include <hip/hip_runtime.h>
#include <hip/hip_bf16.h>

typedef __attribute__((ext_vector_type(16))) __bf16 v16bf;
typedef __attribute__((ext_vector_type(8)))  __bf16 v8bf;
typedef __attribute__((ext_vector_type(8)))  float  v8f;

#define KPAD 2432   // 49*49=2401 padded up to multiple of 32
#define MROWS 8192  // S*N = 8*1024
#define HDIM 384
#define ODIM 256

__device__ __forceinline__ unsigned short f2bf(float f) {
    unsigned int u = __float_as_uint(f);
    unsigned int r = u + 0x7FFFu + ((u >> 16) & 1u);   // RNE
    return (unsigned short)(r >> 16);
}

// A-fragment (16x32 bf16, M x K): lane l<16 -> row M=l, K = {kb..kb+7, kb+16..kb+23}
//                                 lane l>=16 -> row M=l-16, K = {kb+8..kb+15, kb+24..kb+31}
__device__ __forceinline__ v16bf load_fragA(const unsigned short* base, int ld,
                                            int row0, int kb, int lane) {
    int l = lane & 15, hi = lane >> 4;
    const unsigned short* p = base + (size_t)(row0 + l) * ld + kb + hi * 8;
    v8bf lo  = *(const v8bf*)(p);
    v8bf hi8 = *(const v8bf*)(p + 16);
    return __builtin_shufflevector(lo, hi8, 0,1,2,3,4,5,6,7,8,9,10,11,12,13,14,15);
}

// B-fragment (32x16 bf16, K x N), sourced from transposed operand (rows = N, contiguous K):
// lane l<16 -> col N=l, K = {kb..kb+7, kb+16..kb+23} ; lane l>=16 -> col N=l-16, K = {kb+8..kb+15, kb+24..kb+31}
__device__ __forceinline__ v16bf load_fragB(const unsigned short* baseT, int ld,
                                            int col0, int kb, int lane) {
    int l = lane & 15, hi = lane >> 4;
    const unsigned short* p = baseT + (size_t)(col0 + l) * ld + kb + hi * 8;
    v8bf lo  = *(const v8bf*)(p);
    v8bf hi8 = *(const v8bf*)(p + 16);
    return __builtin_shufflevector(lo, hi8, 0,1,2,3,4,5,6,7,8,9,10,11,12,13,14,15);
}

// ---------------- weight pre-transpose + bf16 convert ----------------
__global__ __launch_bounds__(256) void k_convert(const float* __restrict__ w1,
                                                 const float* __restrict__ w2,
                                                 unsigned short* __restrict__ W1T,
                                                 unsigned short* __restrict__ W2T) {
    int i = blockIdx.x * 256 + threadIdx.x;
    const int n1 = HDIM * KPAD;
    if (i < n1) {
        int r = i / KPAD, k = i - r * KPAD;
        float v = (k < 2401) ? w1[(size_t)k * HDIM + r] : 0.0f;
        W1T[i] = f2bf(v);
    } else {
        int j = i - n1;
        if (j < ODIM * HDIM) {
            int r = j / HDIM, k = j - r * HDIM;
            W2T[j] = f2bf(w2[(size_t)k * ODIM + r]);
        }
    }
}

// ---------------- bilinear sample + 49x49 correlation volume (WMMA) ----------------
__global__ __launch_bounds__(256) void k_sample_vol(const float* __restrict__ fmap,
                                                    const float* __restrict__ track,
                                                    const float* __restrict__ coords,
                                                    unsigned short* __restrict__ volB,
                                                    int Hh, int Ww, int lvl) {
    __shared__ __align__(16) unsigned short Ash[64 * 128];
    __shared__ __align__(16) unsigned short Bsh[64 * 128];
    __shared__ int   offs[49][4];     // 4 tap offsets per support point
    __shared__ float wts[49][4];      // 4 tap weights per support point
    int blk = blockIdx.x;             // s*1024 + n
    int s = blk >> 10, n = blk & 1023;
    int tid = threadIdx.x;

    if (tid < 49) {                   // one-time bilinear setup per support point
        float inv = 1.0f / (float)(1 << lvl);
        float cx = coords[(size_t)blk * 2 + 0] * inv;
        float cy = coords[(size_t)blk * 2 + 1] * inv;
        int a = tid / 7, b = tid - a * 7;
        float x = cx + (float)(a - 3);
        float y = cy + (float)(b - 3);
        float x0f = floorf(x), y0f = floorf(y);
        float wx = x - x0f, wy = y - y0f;
        int x0 = min(max((int)x0f, 0), Ww - 1);
        int x1 = min(max((int)x0f + 1, 0), Ww - 1);
        int y0 = min(max((int)y0f, 0), Hh - 1);
        int y1 = min(max((int)y0f + 1, 0), Hh - 1);
        offs[tid][0] = y0 * Ww + x0;  wts[tid][0] = (1.f - wx) * (1.f - wy);
        offs[tid][1] = y0 * Ww + x1;  wts[tid][1] = wx * (1.f - wy);
        offs[tid][2] = y1 * Ww + x0;  wts[tid][2] = (1.f - wx) * wy;
        offs[tid][3] = y1 * Ww + x1;  wts[tid][3] = wx * wy;
    }
    __syncthreads();

    // corr_feat tile: rows p = hw grid point (a*7+b), cols c ; pad rows -> 0
    int HW = Hh * Ww;
    for (int e = tid; e < 64 * 128; e += 256) {
        int p = e >> 7, c = e & 127;
        float v = 0.0f;
        if (p < 49) {
            const float* fp = fmap + ((size_t)s * 128 + c) * HW;
            v = wts[p][0] * fp[offs[p][0]] + wts[p][1] * fp[offs[p][1]]
              + wts[p][2] * fp[offs[p][2]] + wts[p][3] * fp[offs[p][3]];
        }
        Ash[e] = f2bf(v);
    }
    // track tile: rows ij, cols c ; track layout (G*G, N, C) ; pad rows -> 0
    for (int e = tid; e < 64 * 128; e += 256) {
        int ij = e >> 7, c = e & 127;
        float v = (ij < 49) ? track[((size_t)ij * 1024 + n) * 128 + c] : 0.0f;
        Bsh[e] = f2bf(v);
    }
    __syncthreads();

    int wv = tid >> 5, lane = tid & 31;
    v8f accT[2];
#pragma unroll
    for (int tt = 0; tt < 2; ++tt) {               // 16 tiles over 8 waves
        int t = wv * 2 + tt;
        int mt = t >> 2, nt = t & 3;
        v8f acc = (v8f)(0.0f);
#pragma unroll
        for (int ks = 0; ks < 128; ks += 32) {
            v16bf a = load_fragA(Ash, 128, mt * 16, ks, lane);
            v16bf b = load_fragB(Bsh, 128, nt * 16, ks, lane);
            acc = __builtin_amdgcn_wmma_f32_16x16x32_bf16(false, a, false, b,
                                                          (short)0, acc, false, false);
        }
        accT[tt] = acc;
    }
    __syncthreads();                               // all LDS reads done
    // stage full 64x64 D tile (padding region is exactly zero) into Ash, ld=64
    {
        int hi = lane >> 4, l = lane & 15;
#pragma unroll
        for (int tt = 0; tt < 2; ++tt) {
            int t = wv * 2 + tt;
            int mt = t >> 2, nt = t & 3;
            int col = nt * 16 + l;
#pragma unroll
            for (int v = 0; v < 8; ++v)
                Ash[(mt * 16 + v + hi * 8) * 64 + col] = f2bf(accT[tt][v]);
        }
    }
    __syncthreads();
    // coalesced, branch-free copy of the 2432-wide padded vol row (pads are zeros)
    unsigned short* volRow = volB + (size_t)blk * KPAD;
    for (int e = tid; e < KPAD; e += 256) {
        int m = e / 49, c = e - m * 49;            // m==49 row of staging is zero
        volRow[e] = Ash[m * 64 + c];
    }
}

// ---------------- GEMM1: vol(8192x2432) @ w1 -> gelu -> H(8192x384) bf16 ----------------
// one wave computes a 32x64 output tile: 8 WMMA per K-step from 6 fragment loads
__global__ __launch_bounds__(256) void k_gemm1(const unsigned short* __restrict__ X,
                                               const unsigned short* __restrict__ W1T,
                                               const float* __restrict__ b1,
                                               unsigned short* __restrict__ Hb) {
    int tid = threadIdx.x;
    int wgid = blockIdx.x * 8 + (tid >> 5);   // 1536 waves = 256 m32-tiles * 6 nstrips
    int lane = tid & 31;
    int mt = wgid / 6, ns = wgid - mt * 6;
    int m0 = mt * 32, n0 = ns * 64;
    v8f acc[2][4];
#pragma unroll
    for (int mi = 0; mi < 2; ++mi)
#pragma unroll
        for (int j = 0; j < 4; ++j) acc[mi][j] = (v8f)(0.0f);
    for (int k = 0; k < KPAD; k += 32) {
        v16bf a0 = load_fragA(X, KPAD, m0,      k, lane);
        v16bf a1 = load_fragA(X, KPAD, m0 + 16, k, lane);
#pragma unroll
        for (int j = 0; j < 4; ++j) {
            v16bf b = load_fragB(W1T, KPAD, n0 + j * 16, k, lane);
            acc[0][j] = __builtin_amdgcn_wmma_f32_16x16x32_bf16(false, a0, false, b,
                                                                (short)0, acc[0][j], false, false);
            acc[1][j] = __builtin_amdgcn_wmma_f32_16x16x32_bf16(false, a1, false, b,
                                                                (short)0, acc[1][j], false, false);
        }
    }
    int hi = lane >> 4, l = lane & 15;
#pragma unroll
    for (int j = 0; j < 4; ++j) {
        int col = n0 + j * 16 + l;
        float bb = b1[col];
#pragma unroll
        for (int mi = 0; mi < 2; ++mi) {
#pragma unroll
            for (int v = 0; v < 8; ++v) {
                int m = m0 + mi * 16 + v + hi * 8;
                float x = acc[mi][j][v] + bb;
                float g = 0.5f * x * (1.0f + erff(x * 0.70710678118654752f));  // exact gelu
                Hb[(size_t)m * HDIM + col] = f2bf(g);
            }
        }
    }
}

// ---------------- GEMM2: H(8192x384) @ w2 + b2 + time_emb -> out channels ----------------
__global__ __launch_bounds__(256) void k_gemm2(const unsigned short* __restrict__ Hb,
                                               const unsigned short* __restrict__ W2T,
                                               const float* __restrict__ b2,
                                               const float* __restrict__ te,   // (8,1110)
                                               float* __restrict__ out, int lvl) {
    int tid = threadIdx.x;
    int wgid = blockIdx.x * 8 + (tid >> 5);   // 1024 waves = 256 m32-tiles * 4 nstrips
    int lane = tid & 31;
    int mt = wgid >> 2, ns = wgid & 3;
    int m0 = mt * 32, n0 = ns * 64;
    v8f acc[2][4];
#pragma unroll
    for (int mi = 0; mi < 2; ++mi)
#pragma unroll
        for (int j = 0; j < 4; ++j) acc[mi][j] = (v8f)(0.0f);
#pragma unroll 2
    for (int k = 0; k < HDIM; k += 32) {
        v16bf a0 = load_fragA(Hb, HDIM, m0,      k, lane);
        v16bf a1 = load_fragA(Hb, HDIM, m0 + 16, k, lane);
#pragma unroll
        for (int j = 0; j < 4; ++j) {
            v16bf b = load_fragB(W2T, HDIM, n0 + j * 16, k, lane);
            acc[0][j] = __builtin_amdgcn_wmma_f32_16x16x32_bf16(false, a0, false, b,
                                                                (short)0, acc[0][j], false, false);
            acc[1][j] = __builtin_amdgcn_wmma_f32_16x16x32_bf16(false, a1, false, b,
                                                                (short)0, acc[1][j], false, false);
        }
    }
    int hi = lane >> 4, l = lane & 15;
    int ch0 = 2 + lvl * ODIM;
#pragma unroll
    for (int j = 0; j < 4; ++j) {
        int col = n0 + j * 16 + l;
        float bb = b2[col];
#pragma unroll
        for (int mi = 0; mi < 2; ++mi) {
#pragma unroll
            for (int v = 0; v < 8; ++v) {
                int m = m0 + mi * 16 + v + hi * 8;    // m = s*1024 + n
                int s = m >> 10, n = m & 1023;
                float val = acc[mi][j][v] + bb + te[s * 1110 + ch0 + col];
                out[((size_t)n * 8 + s) * 1110 + ch0 + col] = val;
            }
        }
    }
}

// ---------------- tail: vis/conf + posenc(rel coords) + time_emb ----------------
__global__ __launch_bounds__(256) void k_tail(const float* __restrict__ coords,
                                              const float* __restrict__ vis,
                                              const float* __restrict__ conf,
                                              const float* __restrict__ te,
                                              float* __restrict__ out) {
    int i = blockIdx.x * 256 + threadIdx.x;     // 8192 = S*N
    if (i >= MROWS) return;
    int s = i >> 10, n = i & 1023;
    const float* t = te + s * 1110;
    float* o = out + ((size_t)n * 8 + s) * 1110;
    o[0] = vis[i] + t[0];
    o[1] = conf[i] + t[1];
    float cx = coords[(size_t)i * 2], cy = coords[(size_t)i * 2 + 1];
    float fx = 0.f, fy = 0.f, bx = 0.f, by = 0.f;
    if (s < 7) {
        fx = cx - coords[((size_t)(s + 1) * 1024 + n) * 2];
        fy = cy - coords[((size_t)(s + 1) * 1024 + n) * 2 + 1];
    }
    if (s > 0) {
        bx = cx - coords[((size_t)(s - 1) * 1024 + n) * 2];
        by = cy - coords[((size_t)(s - 1) * 1024 + n) * 2 + 1];
    }
    float xv[4] = { fx / 128.f, fy / 96.f, bx / 128.f, by / 96.f };
    float* r = o + 1026;
    const float* tr = t + 1026;
#pragma unroll
    for (int d = 0; d < 4; ++d) r[d] = xv[d] + tr[d];
#pragma unroll
    for (int dg = 0; dg < 10; ++dg) {
        float sc = (float)(1 << dg);
#pragma unroll
        for (int d = 0; d < 4; ++d) {
            float xb = xv[d] * sc;
            r[4 + dg * 4 + d]  = __sinf(xb) + tr[4 + dg * 4 + d];
            r[44 + dg * 4 + d] = __sinf(xb + 1.57079632679489662f) + tr[44 + dg * 4 + d];
        }
    }
}

extern "C" void kernel_launch(void* const* d_in, const int* in_sizes, int n_in,
                              void* d_out, int out_size, void* d_ws, size_t ws_size,
                              hipStream_t stream) {
    (void)in_sizes; (void)n_in; (void)out_size; (void)ws_size;
    const float* fmaps[4]  = { (const float*)d_in[0], (const float*)d_in[1],
                               (const float*)d_in[2], (const float*)d_in[3] };
    const float* tracks[4] = { (const float*)d_in[4], (const float*)d_in[5],
                               (const float*)d_in[6], (const float*)d_in[7] };
    const float* coords = (const float*)d_in[8];
    const float* vis    = (const float*)d_in[9];
    const float* conf   = (const float*)d_in[10];
    const float* w1     = (const float*)d_in[11];
    const float* b1     = (const float*)d_in[12];
    const float* w2     = (const float*)d_in[13];
    const float* b2     = (const float*)d_in[14];
    const float* te     = (const float*)d_in[15];
    float* out = (float*)d_out;

    unsigned short* volB = (unsigned short*)d_ws;                         // 8192*2432
    unsigned short* Hbuf = volB + (size_t)MROWS * KPAD;                   // 8192*384
    unsigned short* W1T  = Hbuf + (size_t)MROWS * HDIM;                   // 384*2432
    unsigned short* W2T  = W1T  + (size_t)HDIM * KPAD;                    // 256*384

    {
        int total = HDIM * KPAD + ODIM * HDIM;
        k_convert<<<(total + 255) / 256, 256, 0, stream>>>(w1, w2, W1T, W2T);
    }
    for (int lvl = 0; lvl < 4; ++lvl) {
        int Hh = 96 >> lvl, Ww = 128 >> lvl;
        k_sample_vol<<<MROWS, 256, 0, stream>>>(fmaps[lvl], tracks[lvl], coords,
                                                volB, Hh, Ww, lvl);
        k_gemm1<<<(256 * 6) / 8, 256, 0, stream>>>(volB, W1T, b1, Hbuf);
        k_gemm2<<<(256 * 4) / 8, 256, 0, stream>>>(Hbuf, W2T, b2, te, out, lvl);
    }
    k_tail<<<MROWS / 256, 256, 0, stream>>>(coords, vis, conf, te, out);
}
